// customConv2_12506944766641
// MI455X (gfx1250) — matrix operand order repaired and checked by reading, not compiled, not run on hardware
//
#include <hip/hip_runtime.h>
#include <hip/hip_bf16.h>

typedef __attribute__((ext_vector_type(16))) _Float16 v16h;
typedef __attribute__((ext_vector_type(8)))  float    v8f;

// Polynomial coefficients (identity branch is exactly zero -> dropped).
#define P01 (-0.000287f)
#define P11 (0.266f)
#define P21 (-0.1097f)

// Problem shape
#define NB   16
#define CH   64     // in = out channels
#define HW   128
#define KSTEPS 18   // K = 9*64 = 576 = 18 * 32

// ---------------------------------------------------------------------------
// Pre-kernel: weights f32 (O=64, I=64, 3, 3) -> f16 in WMMA A-operand lane
// layout, one 1 KiB tile per (kstep t, row-tile mt):
//   wsA[ ((t*4+mt)*32 + lane)*16 + e ]
// A 16x32 f16 layout used here (same k order as B): lane L holds row M=L%16, g=L/16,
//   K(e) = 16g + e
// K-ordering of the GEMM: k = r*64 + ci, r = kh*3+kw; kstep t covers
// r = t/2, ci = 32*(t&1) + K.
// ---------------------------------------------------------------------------
__global__ void wprep_kernel(const float* __restrict__ w, _Float16* __restrict__ wsA)
{
    int idx = blockIdx.x * blockDim.x + threadIdx.x;   // 0 .. 36863
    int e    = idx & 15;
    int lane = (idx >> 4) & 31;
    int tile = idx >> 9;          // 0..71
    int mt   = tile & 3;
    int t    = tile >> 2;         // 0..17

    int g  = lane >> 4;
    int m  = lane & 15;
    int co = mt * 16 + m;
    int K  = 16 * g + e;
    int r  = t >> 1;                      // filter tap 0..8 (kh*3+kw)
    int ci = ((t & 1) << 5) + K;          // 0..63

    wsA[idx] = (_Float16)w[(co * CH + ci) * 9 + r];
}

// ---------------------------------------------------------------------------
// Main kernel: one workgroup = one (image n, output row h).
// Phase 1: stage poly(img) for rows h-1..h+1 into LDS as f16,
//          layout act[row(3)][col(130, incl. zero halo)][ci(64)].
// Phase 2: implicit GEMM 64x128 output tile, K=576, via f16 WMMA.
// ---------------------------------------------------------------------------
__global__ void __launch_bounds__(256)
conv_poly_wmma_kernel(const float* __restrict__ img,
                      const _Float16* __restrict__ wsA,
                      float* __restrict__ out)
{
    __shared__ _Float16 act[3 * 130 * CH];   // 49,920 bytes

    const int n = blockIdx.x >> 7;
    const int h = blockIdx.x & 127;
    const int tid = threadIdx.x;

    // ---- Phase 1: polynomial + halo staging (192 of 256 threads) ----
    if (tid < 3 * CH) {
        const int row = tid >> 6;        // 0..2
        const int ci  = tid & 63;
        const int srcRow = h + row - 1;
        // zero halo columns (x = -1 and x = 128)
        act[(row * 130 + 0)   * CH + ci] = (_Float16)0.0f;
        act[(row * 130 + 129) * CH + ci] = (_Float16)0.0f;
        if (srcRow >= 0 && srcRow < HW) {
            const float* src = img + ((size_t)(n * CH + ci) * HW + srcRow) * HW;
            #pragma unroll 4
            for (int c0 = 0; c0 < HW; c0 += 4) {
                float4 v = *(const float4*)(src + c0);
                float p0 = P01 + v.x * (P11 + P21 * v.x);
                float p1 = P01 + v.y * (P11 + P21 * v.y);
                float p2 = P01 + v.z * (P11 + P21 * v.z);
                float p3 = P01 + v.w * (P11 + P21 * v.w);
                int base = (row * 130 + (c0 + 1)) * CH + ci;
                act[base + 0 * CH] = (_Float16)p0;
                act[base + 1 * CH] = (_Float16)p1;
                act[base + 2 * CH] = (_Float16)p2;
                act[base + 3 * CH] = (_Float16)p3;
            }
        } else {
            for (int c = 1; c <= 128; ++c)
                act[(row * 130 + c) * CH + ci] = (_Float16)0.0f;
        }
    }
    __syncthreads();

    // ---- Phase 2: WMMA implicit GEMM ----
    const int lane = tid & 31;
    const int wv   = tid >> 5;       // wave id = pixel tile nt (0..7)
    const int lg   = lane >> 4;      // K-half selector
    const int lm   = lane & 15;      // col within tile / row within tile

    v8f acc0 = {}, acc1 = {}, acc2 = {}, acc3 = {};

    for (int t = 0; t < KSTEPS; ++t) {
        const int r  = t >> 1;           // filter tap
        const int kh = r / 3;
        const int kw = r % 3;
        const int cibase = (t & 1) << 5;

        // B fragment: 32x16 f16, lane L holds col = 16*wv + lm,
        // K = 16*lg + e -> ci contiguous in LDS (one 32B read per lane).
        const int col = wv * 16 + lm + kw;          // x+1 in [0,129]
        const v16h b = *(const v16h*)&act[(kh * 130 + col) * CH + cibase + (lg << 4)];

        // A fragments: pre-swizzled, 32B/lane coalesced, L2-resident.
        const _Float16* aptr = wsA + (size_t)(t * 4) * 512 + lane * 16;
        v16h a0 = *(const v16h*)(aptr + 0 * 512);
        v16h a1 = *(const v16h*)(aptr + 1 * 512);
        v16h a2 = *(const v16h*)(aptr + 2 * 512);
        v16h a3 = *(const v16h*)(aptr + 3 * 512);

        acc0 = __builtin_amdgcn_wmma_f32_16x16x32_f16(false, a0, false, b, (short)0, acc0, false, false);
        acc1 = __builtin_amdgcn_wmma_f32_16x16x32_f16(false, a1, false, b, (short)0, acc1, false, false);
        acc2 = __builtin_amdgcn_wmma_f32_16x16x32_f16(false, a2, false, b, (short)0, acc2, false, false);
        acc3 = __builtin_amdgcn_wmma_f32_16x16x32_f16(false, a3, false, b, (short)0, acc3, false, false);
    }

    // ---- Epilogue: fold the /27, scatter C tiles (layout ISA 7.12.2) ----
    const float scale = 1.0f / 27.0f;
    const int p = wv * 16 + lm;
    const v8f accs[4] = {acc0, acc1, acc2, acc3};
    #pragma unroll
    for (int mt = 0; mt < 4; ++mt) {
        const int coBase = mt * 16 + lg * 8;
        #pragma unroll
        for (int rr = 0; rr < 8; ++rr) {
            const int co = coBase + rr;
            out[((size_t)(n * CH + co) * HW + h) * HW + p] = accs[mt][rr] * scale;
        }
    }
}

// ---------------------------------------------------------------------------
extern "C" void kernel_launch(void* const* d_in, const int* in_sizes, int n_in,
                              void* d_out, int out_size, void* d_ws, size_t ws_size,
                              hipStream_t stream)
{
    const float* img     = (const float*)d_in[0];
    const float* weights = (const float*)d_in[1];
    // d_in[2] (identity_kernel) contributes exactly zero -> unused.
    float* out = (float*)d_out;
    _Float16* wsA = (_Float16*)d_ws;   // 72 KiB of swizzled f16 weights

    // 18*4 tiles * 32 lanes * 16 elems = 36864 f16
    wprep_kernel<<<144, 256, 0, stream>>>(weights, wsA);

    // one WG per (n, h): 16 * 128 = 2048 WGs, 256 threads (8 waves)
    conv_poly_wmma_kernel<<<NB * HW, 256, 0, stream>>>(img, wsA, out);
}
